// MultiHeadAttention_50397146251544
// MI455X (gfx1250) — matrix operand order repaired and checked by reading, not compiled, not run on hardware
//
#include <hip/hip_runtime.h>
#include <hip/hip_bf16.h>

// ---------------------------------------------------------------------------
// Multi-head attention for MI455X (gfx1250): all matmuls via
// v_wmma_f32_16x16x32_f16 (f16 inputs, f32 accumulate), flash-attention style
// softmax so the B*H*S*S score tensor is never materialized.
// CDNA5 data movement: TDM tensor_load_to_lds for the f16 ctx tile staging in
// the output projection (tracked on TENSORcnt), global_prefetch for K/V
// streaming in the flash loop.
// ---------------------------------------------------------------------------

typedef __attribute__((ext_vector_type(16))) _Float16 v16h;
typedef __attribute__((ext_vector_type(8)))  _Float16 v8h;
typedef __attribute__((ext_vector_type(4)))  _Float16 v4h;
typedef __attribute__((ext_vector_type(8)))  float    v8f;
typedef __attribute__((ext_vector_type(4)))  unsigned int u32x4;
typedef __attribute__((ext_vector_type(8)))  unsigned int u32x8;

constexpr int BATCH = 2;
constexpr int SEQ   = 2048;
constexpr int FEAT  = 1024;
constexpr int HEADS = 16;
constexpr int DK    = 64;              // FEAT / HEADS
constexpr int ROWS  = BATCH * SEQ;     // 4096 flattened tokens

// ---- WMMA helpers ---------------------------------------------------------

__device__ inline v8f wmma16x16x32(v16h a, v16h b, v8f c) {
  // D = A(16x32, f16) * B(32x16, f16) + C(16x16, f32)
  return __builtin_amdgcn_wmma_f32_16x16x32_f16(
      /*neg_a=*/false, a, /*neg_b=*/false, b,
      /*c_mod=*/(short)0, c, /*reuse_a=*/false, /*reuse_b=*/false);
}

// Combine two 16-byte half-vectors into one 16-half fragment register set.
__device__ inline v16h load_frag16(const _Float16* p0, const _Float16* p1) {
  v8h a = *(const v8h*)p0;
  v8h b = *(const v8h*)p1;
  v16h r;
#pragma unroll
  for (int i = 0; i < 8; ++i) { r[i] = a[i]; r[i + 8] = b[i]; }
  return r;
}

// A-matrix fragment (16x32): lane = row M (mod 16), hi = lane/16.
// halves e -> K = e + 16*hi (contiguous)
__device__ inline v16h load_afrag(const _Float16* row, int hi) {
  return load_frag16(row + 16 * hi, row + 16 * hi + 8);
}

// ---- Tensor Data Mover: 2D f16 tile load, global -> LDS -------------------
// D# per CDNA5 ISA (2D: VADDR2/3 = NULL form).
// pad_interval code 3 = 16 DWORDs (one 32-half tile row), pad_amount code 3 =
// 4 DWORDs (8 halves) -> reproduces the LDA = 40-half padded LDS pitch.
__device__ inline void tdm_load_tile_f16(unsigned lds_byte_addr,
                                         const _Float16* gptr,
                                         unsigned tile_w, unsigned tile_h,
                                         unsigned tensor_w, unsigned tensor_h,
                                         unsigned row_stride) {
  const unsigned long long ga = (unsigned long long)(uintptr_t)gptr;
  u32x4 g0;
  g0[0] = 1u;                                        // count=1 (valid), user D#
  g0[1] = lds_byte_addr;                             // lds_addr
  g0[2] = (unsigned)(ga & 0xFFFFFFFFu);              // global_addr[31:0]
  g0[3] = (unsigned)((ga >> 32) & 0x01FFFFFFu)       // global_addr[56:32]
        | (2u << 30);                                // type = 2 ("image")
  u32x8 g1;
  g1[0] = (1u << 16)                                 // data_size = 1 -> 2 bytes
        | (1u << 20)                                 // pad_enable
        | (3u << 22)                                 // pad_interval: 16 DWORDs
        | (3u << 25);                                // pad_amount: 4 DWORDs
  g1[1] = (tensor_w & 0xFFFFu) << 16;                                // dim0 lo
  g1[2] = ((tensor_w >> 16) & 0xFFFFu) | ((tensor_h & 0xFFFFu) << 16);
  g1[3] = ((tensor_h >> 16) & 0xFFFFu) | ((tile_w & 0xFFFFu) << 16); // tile0
  g1[4] = (tile_h & 0xFFFFu);                        // tile_dim1; tile_dim2=0
  g1[5] = row_stride;                                // dim0_stride[31:0]
  g1[6] = 0u;                                        // stride hi bits
  g1[7] = 0u;
  asm volatile("tensor_load_to_lds %0, %1" : : "s"(g0), "s"(g1) : "memory");
}

// ---------------------------------------------------------------------------
// Kernel 1: fused QKV projection.  C[4096x1024] = X * W + b, then scatter into
// head-major f16 buffers.  mode 0 -> Q [B,H,S,dk], 1 -> K [B,H,S,dk],
// 2 -> V transposed [B,H,dk,S] (so PV WMMA B-fragments are contiguous).
// Tile: 128x64, 8 waves, each wave a 32x32 block = 2x2 WMMA fragments.
// (A and B are f32 in HBM -> staged with VALU f32->f16 conversion; TDM cannot
// convert, so the plain VMEM path is the right one here.)
// ---------------------------------------------------------------------------
__global__ __launch_bounds__(256) void qkv_gemm_kernel(
    const float* __restrict__ X,
    const float* __restrict__ Wq, const float* __restrict__ bq,
    const float* __restrict__ Wk, const float* __restrict__ bk,
    const float* __restrict__ Wv, const float* __restrict__ bv,
    _Float16* __restrict__ Qf, _Float16* __restrict__ Kf,
    _Float16* __restrict__ VTf) {
  constexpr int LDA = 32 + 8;   // halves, keeps 16B alignment (80B row pitch)
  __shared__ __align__(16) _Float16 As[128 * LDA];   // 128x32 tile of X (f16)
  __shared__ __align__(16) _Float16 BsT[64 * LDA];   // 64x32: W tile TRANSPOSED

  const int mode = blockIdx.z;
  const float* W    = (mode == 0) ? Wq : (mode == 1) ? Wk : Wv;
  const float* bias = (mode == 0) ? bq : (mode == 1) ? bk : bv;

  const int tid  = threadIdx.x;
  const int lane = tid & 31;
  const int wave = tid >> 5;
  const int wm = wave >> 1;        // 0..3  (rows)
  const int wn = wave & 1;         // 0..1  (cols)
  const int hi  = lane >> 4;
  const int l16 = lane & 15;
  const int m0 = blockIdx.y * 128;
  const int n0 = blockIdx.x * 64;

  const v8f vzero = {0.f, 0.f, 0.f, 0.f, 0.f, 0.f, 0.f, 0.f};
  v8f acc[2][2];
#pragma unroll
  for (int i = 0; i < 2; ++i)
#pragma unroll
    for (int j = 0; j < 2; ++j) acc[i][j] = vzero;

  for (int k0 = 0; k0 < FEAT; k0 += 32) {
    __syncthreads();
    // Stage A tile: 128x32 f32 -> f16. 1024 float4 / 256 threads = 4 each.
#pragma unroll
    for (int i = 0; i < 4; ++i) {
      const int id  = tid + i * 256;
      const int row = id >> 3;
      const int c4  = id & 7;
      const float4 xv =
          *(const float4*)(X + (size_t)(m0 + row) * FEAT + k0 + c4 * 4);
      v4h h4;
      h4[0] = (_Float16)xv.x; h4[1] = (_Float16)xv.y;
      h4[2] = (_Float16)xv.z; h4[3] = (_Float16)xv.w;
      *(v4h*)(As + row * LDA + c4 * 4) = h4;
    }
    // Stage B tile transposed: 32x64 f32 -> BsT[n][k] f16.
#pragma unroll
    for (int i = 0; i < 2; ++i) {
      const int id   = tid + i * 256;
      const int krow = id >> 4;
      const int c4   = id & 15;
      const float4 wv =
          *(const float4*)(W + (size_t)(k0 + krow) * FEAT + n0 + c4 * 4);
      BsT[(c4 * 4 + 0) * LDA + krow] = (_Float16)wv.x;
      BsT[(c4 * 4 + 1) * LDA + krow] = (_Float16)wv.y;
      BsT[(c4 * 4 + 2) * LDA + krow] = (_Float16)wv.z;
      BsT[(c4 * 4 + 3) * LDA + krow] = (_Float16)wv.w;
    }
    __syncthreads();

    v16h a0 = load_afrag(As + (wm * 32 + l16) * LDA, hi);
    v16h a1 = load_afrag(As + (wm * 32 + 16 + l16) * LDA, hi);
    // B fragment: lane = column N, halves K = e + 16*hi (contiguous in BsT).
    const _Float16* bp0 = BsT + (wn * 32 + l16) * LDA + 16 * hi;
    const _Float16* bp1 = BsT + (wn * 32 + 16 + l16) * LDA + 16 * hi;
    v16h b0 = load_frag16(bp0, bp0 + 8);
    v16h b1 = load_frag16(bp1, bp1 + 8);

    acc[0][0] = wmma16x16x32(a0, b0, acc[0][0]);
    acc[0][1] = wmma16x16x32(a0, b1, acc[0][1]);
    acc[1][0] = wmma16x16x32(a1, b0, acc[1][0]);
    acc[1][1] = wmma16x16x32(a1, b1, acc[1][1]);
  }

  // Epilogue: bias add + head-major scatter as f16.
#pragma unroll
  for (int mf = 0; mf < 2; ++mf) {
#pragma unroll
    for (int nf = 0; nf < 2; ++nf) {
      const v8f c = acc[mf][nf];
      const int ncol  = n0 + wn * 32 + nf * 16 + l16;   // 0..1023
      const float bn  = bias[ncol];
      const int mbase = m0 + wm * 32 + mf * 16 + 8 * hi; // rows mbase..mbase+7
      const int bb = mbase >> 11;          // batch (128-row tiles never split)
      const int ss = mbase & (SEQ - 1);
      const int hh = ncol >> 6;
      const int dd = ncol & (DK - 1);
      if (mode == 2) {
        v8h pack;
#pragma unroll
        for (int r = 0; r < 8; ++r) pack[r] = (_Float16)(c[r] + bn);
        *(v8h*)(VTf + (((size_t)bb * HEADS + hh) * DK + dd) * SEQ + ss) = pack;
      } else {
        _Float16* dst = (mode == 0) ? Qf : Kf;
#pragma unroll
        for (int r = 0; r < 8; ++r)
          dst[(((size_t)bb * HEADS + hh) * SEQ + ss + r) * DK + dd] =
              (_Float16)(c[r] + bn);
      }
    }
  }
}

// ---------------------------------------------------------------------------
// Kernel 2: flash attention.  Grid (S/64, B*H); 4 independent waves per WG,
// each wave owns 16 query rows and streams keys in steps of 32.
// Per step: 4 WMMAs (scores, chained over dk=64) + online softmax
// (cross-lane reductions within the 16-lane C-fragment groups) + LDS bounce
// of P into A-layout + 4 WMMAs (P*V over the transposed V).
// K/V are L2-resident (512KB per (b,h) vs 192MB L2); prefetch pulls the next
// 32-key tile toward the WGP one iteration ahead.
// ---------------------------------------------------------------------------
__global__ __launch_bounds__(128) void flash_attn_kernel(
    const _Float16* __restrict__ Qf, const _Float16* __restrict__ Kf,
    const _Float16* __restrict__ VTf, _Float16* __restrict__ ctx) {
  __shared__ __align__(16) _Float16 Plds[4][16][32];   // per-wave P tile

  const int lane = threadIdx.x & 31;
  const int wave = threadIdx.x >> 5;
  const int hi  = lane >> 4;
  const int l16 = lane & 15;
  const int bh  = blockIdx.y;                 // b*HEADS + h
  const int q0  = blockIdx.x * 64 + wave * 16;
  const float scale = 0.125f;                 // 1/sqrt(64)

  const _Float16* Qb = Qf + ((size_t)bh * SEQ + q0) * DK;
  const _Float16* Kb = Kf + (size_t)bh * SEQ * DK;
  const _Float16* Vb = VTf + (size_t)bh * DK * SEQ;    // [dk][SEQ]

  // Q A-fragments (fixed for the whole key loop).
  const _Float16* qrow = Qb + (size_t)l16 * DK;
  const v16h aq0 = load_afrag(qrow, hi);        // dk 0..31
  const v16h aq1 = load_afrag(qrow + 32, hi);   // dk 32..63

  const v8f vzero = {0.f, 0.f, 0.f, 0.f, 0.f, 0.f, 0.f, 0.f};
  v8f o[4];
#pragma unroll
  for (int n = 0; n < 4; ++n) o[n] = vzero;
  float m_r[8], l_r[8];
#pragma unroll
  for (int r = 0; r < 8; ++r) { m_r[r] = -1e30f; l_r[r] = 0.f; }

  for (int kt = 0; kt < SEQ; kt += 32) {
    // Prefetch next key/value tiles (speculative; OOB prefetch is dropped).
    __builtin_prefetch(Kb + (size_t)(kt + 32 + l16) * DK, 0, 1);
    __builtin_prefetch(Vb + (size_t)(lane * 2) * SEQ + kt + 32, 0, 1);

    // ---- scores: 16 queries x 32 keys, K-dim = dk = 64 ----
    const _Float16* k0p = Kb + (size_t)(kt + l16) * DK + 16 * hi;
    const _Float16* k1p = Kb + (size_t)(kt + 16 + l16) * DK + 16 * hi;
    v16h bk00 = load_frag16(k0p, k0p + 8);             // keys kt..+15, dk 0..31
    v16h bk01 = load_frag16(k0p + 32, k0p + 40);       // keys kt..+15, dk 32..63
    v16h bk10 = load_frag16(k1p, k1p + 8);             // keys kt+16..+31
    v16h bk11 = load_frag16(k1p + 32, k1p + 40);

    v8f c0 = wmma16x16x32(aq0, bk00, vzero);
    c0     = wmma16x16x32(aq1, bk01, c0);
    v8f c1 = wmma16x16x32(aq0, bk10, vzero);
    c1     = wmma16x16x32(aq1, bk11, c1);

    // ---- online softmax over the 32 keys of this step ----
    float alpha[8];
#pragma unroll
    for (int r = 0; r < 8; ++r) {
      const float s0 = c0[r] * scale;
      const float s1 = c1[r] * scale;
      float mx = fmaxf(s0, s1);
      mx = fmaxf(mx, __shfl_xor(mx, 1));
      mx = fmaxf(mx, __shfl_xor(mx, 2));
      mx = fmaxf(mx, __shfl_xor(mx, 4));
      mx = fmaxf(mx, __shfl_xor(mx, 8));
      const float nm = fmaxf(m_r[r], mx);
      const float al = __expf(m_r[r] - nm);
      const float p0 = __expf(s0 - nm);
      const float p1 = __expf(s1 - nm);
      float rs = p0 + p1;
      rs += __shfl_xor(rs, 1);
      rs += __shfl_xor(rs, 2);
      rs += __shfl_xor(rs, 4);
      rs += __shfl_xor(rs, 8);
      l_r[r] = l_r[r] * al + rs;
      m_r[r] = nm;
      alpha[r] = al;
      // C-layout (lane = key column) -> row-major P tile in LDS.
      Plds[wave][r + 8 * hi][l16]      = (_Float16)p0;
      Plds[wave][r + 8 * hi][16 + l16] = (_Float16)p1;
    }
#pragma unroll
    for (int n = 0; n < 4; ++n)
#pragma unroll
      for (int r = 0; r < 8; ++r) o[n][r] *= alpha[r];

    // Re-read P in A-fragment layout (16x32, K-dim = 32 keys).
    const _Float16* prow = &Plds[wave][l16][0];
    v16h pa = load_afrag(prow, hi);

    // ---- accumulate O += P * V over the 4 dk column tiles ----
#pragma unroll
    for (int n = 0; n < 4; ++n) {
      const _Float16* vp =
          Vb + (size_t)(n * 16 + l16) * SEQ + kt + 16 * hi;
      v16h vb = load_frag16(vp, vp + 8);
      o[n] = wmma16x16x32(pa, vb, o[n]);
    }
  }

  // ---- normalize and write ctx (f16, [B,S,FEAT]) ----
  const int b = bh / HEADS;
  const int h = bh % HEADS;
  float inv[8];
#pragma unroll
  for (int r = 0; r < 8; ++r) inv[r] = 1.0f / l_r[r];
#pragma unroll
  for (int n = 0; n < 4; ++n) {
    const int col = h * DK + n * 16 + l16;
#pragma unroll
    for (int r = 0; r < 8; ++r) {
      const int srow = q0 + r + 8 * hi;
      ctx[((size_t)b * SEQ + srow) * FEAT + col] = (_Float16)(o[n][r] * inv[r]);
    }
  }
}

// ---------------------------------------------------------------------------
// Kernel 3: output projection.  out[4096x1024] = ctx(f16) * Wo + bo  (f32 out)
// Same 128x64 tiling as kernel 1.  The A tile (ctx) is already f16, so it is
// staged by the Tensor Data Mover: one tensor_load_to_lds per K-step (wave 0
// issues it; TENSORcnt completion; the D# pad fields reproduce the padded
// 40-half LDS pitch), overlapping the VALU f32->f16 conversion of the B tile.
// ---------------------------------------------------------------------------
__global__ __launch_bounds__(256) void out_proj_kernel(
    const _Float16* __restrict__ ctx, const float* __restrict__ Wo,
    const float* __restrict__ bo, float* __restrict__ out) {
  constexpr int LDA = 32 + 8;
  __shared__ __align__(16) _Float16 As[128 * LDA];
  __shared__ __align__(16) _Float16 BsT[64 * LDA];

  const int tid  = threadIdx.x;
  const int lane = tid & 31;
  const int wave = tid >> 5;
  const int wm = wave >> 1;
  const int wn = wave & 1;
  const int hi  = lane >> 4;
  const int l16 = lane & 15;
  const int m0 = blockIdx.y * 128;
  const int n0 = blockIdx.x * 64;

  // Low 32 bits of the generic address of a __shared__ object are its LDS
  // byte offset (ISA §10.2 LDS aperture).
  const unsigned As_lds = (unsigned)(uintptr_t)(As);

  const v8f vzero = {0.f, 0.f, 0.f, 0.f, 0.f, 0.f, 0.f, 0.f};
  v8f acc[2][2];
#pragma unroll
  for (int i = 0; i < 2; ++i)
#pragma unroll
    for (int j = 0; j < 2; ++j) acc[i][j] = vzero;

  for (int k0 = 0; k0 < FEAT; k0 += 32) {
    __syncthreads();
    // Stage A via TDM: 128x32 f16 tile, row pitch FEAT halves, LDS pitch
    // padded to 40 halves by the descriptor's pad fields. Wave 0 issues.
    if (wave == 0) {
      tdm_load_tile_f16(As_lds, ctx + (size_t)m0 * FEAT + k0,
                        /*tile_w=*/32, /*tile_h=*/128,
                        /*tensor_w=*/FEAT, /*tensor_h=*/ROWS,
                        /*row_stride=*/FEAT);
    }
    // Stage B tile transposed while the TDM runs: 32x64 f32 -> BsT[n][k] f16.
#pragma unroll
    for (int i = 0; i < 2; ++i) {
      const int id   = tid + i * 256;
      const int krow = id >> 4;
      const int c4   = id & 15;
      const float4 wv =
          *(const float4*)(Wo + (size_t)(k0 + krow) * FEAT + n0 + c4 * 4);
      BsT[(c4 * 4 + 0) * LDA + krow] = (_Float16)wv.x;
      BsT[(c4 * 4 + 1) * LDA + krow] = (_Float16)wv.y;
      BsT[(c4 * 4 + 2) * LDA + krow] = (_Float16)wv.z;
      BsT[(c4 * 4 + 3) * LDA + krow] = (_Float16)wv.w;
    }
    if (wave == 0) __builtin_amdgcn_s_wait_tensorcnt(0);
    __syncthreads();

    v16h a0 = load_afrag(As + (wm * 32 + l16) * LDA, hi);
    v16h a1 = load_afrag(As + (wm * 32 + 16 + l16) * LDA, hi);
    const _Float16* bp0 = BsT + (wn * 32 + l16) * LDA + 16 * hi;
    const _Float16* bp1 = BsT + (wn * 32 + 16 + l16) * LDA + 16 * hi;
    v16h b0 = load_frag16(bp0, bp0 + 8);
    v16h b1 = load_frag16(bp1, bp1 + 8);

    acc[0][0] = wmma16x16x32(a0, b0, acc[0][0]);
    acc[0][1] = wmma16x16x32(a0, b1, acc[0][1]);
    acc[1][0] = wmma16x16x32(a1, b0, acc[1][0]);
    acc[1][1] = wmma16x16x32(a1, b1, acc[1][1]);
  }

#pragma unroll
  for (int mf = 0; mf < 2; ++mf) {
#pragma unroll
    for (int nf = 0; nf < 2; ++nf) {
      const v8f c = acc[mf][nf];
      const int ncol  = n0 + wn * 32 + nf * 16 + l16;
      const float bn  = bo[ncol];
      const int mbase = m0 + wm * 32 + mf * 16 + 8 * hi;
#pragma unroll
      for (int r = 0; r < 8; ++r)
        out[(size_t)(mbase + r) * FEAT + ncol] = c[r] + bn;
    }
  }
}

// ---------------------------------------------------------------------------
// Launcher.  Workspace layout (f16): Q | K | V^T | ctx  = 4 x 8 MiB = 32 MiB.
// ---------------------------------------------------------------------------
extern "C" void kernel_launch(void* const* d_in, const int* in_sizes, int n_in,
                              void* d_out, int out_size, void* d_ws,
                              size_t ws_size, hipStream_t stream) {
  const float* x  = (const float*)d_in[0];
  const float* Wq = (const float*)d_in[1];
  const float* bq = (const float*)d_in[2];
  const float* Wk = (const float*)d_in[3];
  const float* bk = (const float*)d_in[4];
  const float* Wv = (const float*)d_in[5];
  const float* bv = (const float*)d_in[6];
  const float* Wo = (const float*)d_in[7];
  const float* bo = (const float*)d_in[8];
  float* out = (float*)d_out;

  const size_t elems = (size_t)ROWS * FEAT;   // 4096*1024
  _Float16* Qf  = (_Float16*)d_ws;
  _Float16* Kf  = Qf + elems;
  _Float16* VTf = Kf + elems;
  _Float16* ctx = VTf + elems;

  // 1) fused QKV projection (z = 0:Q, 1:K, 2:V-transposed)
  qkv_gemm_kernel<<<dim3(FEAT / 64, ROWS / 128, 3), 256, 0, stream>>>(
      x, Wq, bq, Wk, bk, Wv, bv, Qf, Kf, VTf);

  // 2) flash attention: 64 query rows per WG (4 waves x 16 rows)
  flash_attn_kernel<<<dim3(SEQ / 64, BATCH * HEADS), 128, 0, stream>>>(
      Qf, Kf, VTf, ctx);

  // 3) output projection
  out_proj_kernel<<<dim3(FEAT / 64, ROWS / 128, 1), 256, 0, stream>>>(
      ctx, Wo, bo, out);
}
